// MultiHeadAttention_58506044506743
// MI455X (gfx1250) — matrix operand order repaired and checked by reading, not compiled, not run on hardware
//
#include <hip/hip_runtime.h>
#include <hip/hip_bf16.h>
#include <cmath>

// Problem constants (match reference)
#define BB 2
#define LL 2048
#define DIN 1024
#define HH 16
#define KD 64
#define MM (BB * LL)          // 4096 rows
#define SCALE_INV 0.125f      // 1 / sqrt(IN_DIM/H) = 1/8

typedef __attribute__((ext_vector_type(16))) _Float16 v16h;
typedef __attribute__((ext_vector_type(8)))  float    v8f;

// ---------------------------------------------------------------------------
// WMMA helper: D = A(16x32 f16) * B(32x16 f16) + C(16x16 f32)
// ---------------------------------------------------------------------------
__device__ __forceinline__ v8f wmma_f16(v16h a, v16h b, v8f c) {
    return __builtin_amdgcn_wmma_f32_16x16x32_f16(
        /*neg_a=*/false, a, /*neg_b=*/false, b,
        /*c_mod=*/(short)0, c, /*reuse_a=*/false, /*reuse_b=*/false);
}

// A-fragment (16x32, f16, MxK): lanes 0-15 row M=lane hold K 0..15,
// lanes 16-31 row M=lane-16 hold K 16..31.  src = row-major [16][stride]
__device__ __forceinline__ v16h load_a_frag_h(const _Float16* base, int stride, int lane) {
    int r  = lane & 15;
    int b0 = (lane & 16) ? 16 : 0;
    const _Float16* p = base + (size_t)r * stride + b0;
    v16h a;
#pragma unroll
    for (int i = 0; i < 8; ++i) a[i]     = p[i];
#pragma unroll
    for (int i = 0; i < 8; ++i) a[8 + i] = p[8 + i];
    return a;
}

// Same A layout, f32 source converted on load (atn probs in LDS, f32 inputs).
__device__ __forceinline__ v16h load_a_frag_f32(const float* base, int stride, int lane) {
    int r  = lane & 15;
    int b0 = (lane & 16) ? 16 : 0;
    const float* p = base + (size_t)r * stride + b0;
    v16h a;
#pragma unroll
    for (int i = 0; i < 8; ++i) a[i]     = (_Float16)p[i];
#pragma unroll
    for (int i = 0; i < 8; ++i) a[8 + i] = (_Float16)p[8 + i];
    return a;
}

// B-fragment (32x16, f16, KxN): lane n (0-15) = col N=n, K=0..15; lanes 16-31
// col N=lane-16, K=16..31.  Source is K-contiguous [n][k]: 32B per lane.
__device__ __forceinline__ v16h load_b_frag_h(const _Float16* base, int stride, int lane) {
    int n   = lane & 15;
    int kh2 = (lane >> 4) << 4;
    const _Float16* p = base + (size_t)n * stride + kh2;
    v16h b;
#pragma unroll
    for (int i = 0; i < 16; ++i) b[i] = p[i];
    return b;
}

// ---------------------------------------------------------------------------
// Weight pre-pass: WT[n][k] = (f16) W[k][n]   (1024x1024, LDS tile transpose)
// ---------------------------------------------------------------------------
__global__ __launch_bounds__(256) void wtrans_kernel(const float* __restrict__ W,
                                                     _Float16* __restrict__ WT) {
    __shared__ float tile[64][65];
    const int k0 = blockIdx.y * 64, n0 = blockIdx.x * 64;
    const int tx = threadIdx.x & 63, tg = threadIdx.x >> 6;  // 4 row groups
#pragma unroll
    for (int i = 0; i < 16; ++i) {
        int r = tg * 16 + i;
        tile[r][tx] = W[(size_t)(k0 + r) * DIN + n0 + tx];
    }
    __syncthreads();
#pragma unroll
    for (int i = 0; i < 16; ++i) {
        int r = tg * 16 + i;
        WT[(size_t)(n0 + r) * DIN + k0 + tx] = (_Float16)tile[tx][r];
    }
}

// ---------------------------------------------------------------------------
// LDS-free streaming GEMM: C[4096,1024] = A[4096,1024] x WT[1024,1024]^T.
// Working set lives in the 192MB L2, so fragments stream straight from global
// (per-lane-contiguous b128 loads), no staging, no barriers.
// Wave tile 32x64 (2x4 WMMA tiles); block = 8 waves -> 64x256 block tile.
// MODE 0: out f16 head-split qh/kh[z][l][d]
// MODE 2: out f16 transposed vhT[z][d][l]
// MODE 1: out f32 = C + bias + residual (pre-LayerNorm x)
// ---------------------------------------------------------------------------
template <int MODE, typename AT>
__global__ __launch_bounds__(256) void gemm_kernel(const AT* __restrict__ A,
                                                   const _Float16* __restrict__ WT,
                                                   const float* __restrict__ bias,
                                                   const float* __restrict__ resid,
                                                   _Float16* __restrict__ outH,
                                                   float* __restrict__ outF) {
    const int t    = threadIdx.x;
    const int wave = t >> 5;
    const int lane = t & 31;
    const int bm = blockIdx.y * 64  + (wave >> 2) * 32;   // wave row origin
    const int bn = blockIdx.x * 256 + (wave & 3) * 64;    // wave col origin

    const int r   = lane & 15;
    const int b0  = (lane & 16) ? 16 : 0;
    const int kh2 = (lane >> 4) << 4;
    const AT* a0p = A + (size_t)(bm + r) * DIN + b0;        // row tile 0
    const AT* a1p = A + (size_t)(bm + 16 + r) * DIN + b0;   // row tile 1
    const _Float16* bp = WT + (size_t)(bn + r) * DIN + kh2; // col base (r = n)

    v8f cf[2][4] = {};

    for (int kt = 0; kt < 1024; kt += 32) {
        v16h a0, a1;
        {
            const AT* p = a0p + kt;
#pragma unroll
            for (int i = 0; i < 8; ++i) { a0[i] = (_Float16)p[i]; a0[8 + i] = (_Float16)p[8 + i]; }
            p = a1p + kt;
#pragma unroll
            for (int i = 0; i < 8; ++i) { a1[i] = (_Float16)p[i]; a1[8 + i] = (_Float16)p[8 + i]; }
        }
        if (kt + 32 < 1024) {
            __builtin_prefetch(a0p + kt + 32, 0, 0);
            __builtin_prefetch(bp + kt + 32, 0, 0);
        }
#pragma unroll
        for (int ct = 0; ct < 4; ++ct) {
            const _Float16* p = bp + (size_t)ct * 16 * DIN + kt;
            v16h b;
#pragma unroll
            for (int i = 0; i < 16; ++i) b[i] = p[i];
            cf[0][ct] = wmma_f16(a0, b, cf[0][ct]);
            cf[1][ct] = wmma_f16(a1, b, cf[1][ct]);
        }
    }

    // Epilogue. C layout: element j -> row = j + 8*(lane>>4), col = lane&15.
    const int ncol = lane & 15;
    const int rofs = (lane >> 4) << 3;
#pragma unroll
    for (int rt = 0; rt < 2; ++rt)
#pragma unroll
        for (int ct = 0; ct < 4; ++ct) {
            v8f c = cf[rt][ct];
            int gcol = bn + ct * 16 + ncol;
            float bb = bias[gcol];
#pragma unroll
            for (int j = 0; j < 8; ++j) {
                int grow  = bm + rt * 16 + j + rofs;
                float val = c[j] + bb;
                if constexpr (MODE == 0) {           // head-split f16 (Q, K)
                    int batch = grow >> 11, l = grow & 2047;
                    int h = gcol >> 6,     d = gcol & 63;
                    int z = h * BB + batch;
                    outH[((size_t)z * LL + l) * KD + d] = (_Float16)val;
                } else if constexpr (MODE == 2) {    // transposed f16 (V)
                    int batch = grow >> 11, l = grow & 2047;
                    int h = gcol >> 6,     d = gcol & 63;
                    int z = h * BB + batch;
                    outH[((size_t)z * KD + d) * LL + l] = (_Float16)val;
                } else {                              // O-proj + residual, f32
                    val += resid[(size_t)grow * DIN + gcol];
                    outF[(size_t)grow * DIN + gcol] = val;
                }
            }
        }
}

// ---------------------------------------------------------------------------
// Fused attention: scores (WMMA) -> softmax (128KB LDS strip) -> atn written
// once to HBM -> P@V (WMMA from LDS, V pre-transposed). One WG per (z, 16 q).
// ---------------------------------------------------------------------------
__global__ __launch_bounds__(256) void attn_kernel(const _Float16* __restrict__ qh,
                                                   const _Float16* __restrict__ kh,
                                                   const _Float16* __restrict__ vhT,
                                                   float* __restrict__ atn_out,
                                                   _Float16* __restrict__ yout) {
    extern __shared__ char smem[];
    float* sS   = (float*)smem;          // 16 x 2048 scores / probs
    float* sRed = sS + 16 * LL;          // 16 x 16 reduction
    float* sMax = sRed + 256;            // 16
    float* sSum = sMax + 16;             // 16
    float* sPar = sSum + 16;             // 4 x 256 partial C for k-split reduce

    const int z    = blockIdx.y;
    const int qb   = blockIdx.x * 16;
    const int wave = threadIdx.x >> 5;
    const int lane = threadIdx.x & 31;

    // ---- Phase 1: S = (Q Kᵀ) / 8 ----------------------------------------
    const _Float16* qbase = qh + ((size_t)z * LL + qb) * KD;
    v16h qa0 = load_a_frag_h(qbase,      KD, lane);   // d = 0..31
    v16h qa1 = load_a_frag_h(qbase + 32, KD, lane);   // d = 32..63
    const int ncol = lane & 15;
    const int rofs = (lane >> 4) << 3;
    for (int it = 0; it < 16; ++it) {
        int kt = it * 128 + wave * 16;   // each wave: its own 16 K-columns
        const _Float16* kb = kh + ((size_t)z * LL + kt) * KD;
        v16h b0 = load_b_frag_h(kb,      KD, lane);   // kh rows are B columns
        v16h b1 = load_b_frag_h(kb + 32, KD, lane);
        v8f c = {};
        c = wmma_f16(qa0, b0, c);
        c = wmma_f16(qa1, b1, c);
#pragma unroll
        for (int j = 0; j < 8; ++j)
            sS[(j + rofs) * LL + kt + ncol] = c[j] * SCALE_INV;
    }
    __syncthreads();

    // ---- Phase 2: softmax over 2048 cols, write atn to HBM ---------------
    const int row = threadIdx.x >> 4;    // 16 rows x 16 threads
    const int sub = threadIdx.x & 15;
    float* srow = sS + (size_t)row * LL;

    float m = -INFINITY;
    for (int c0 = sub; c0 < LL; c0 += 16) m = fmaxf(m, srow[c0]);
    sRed[row * 16 + sub] = m;
    __syncthreads();
    if (sub == 0) {
        float mx = sRed[row * 16];
#pragma unroll
        for (int i = 1; i < 16; ++i) mx = fmaxf(mx, sRed[row * 16 + i]);
        sMax[row] = mx;
    }
    __syncthreads();
    float mx = sMax[row];
    float sum = 0.f;
    for (int c0 = sub; c0 < LL; c0 += 16) {
        float e = __expf(srow[c0] - mx);
        srow[c0] = e;
        sum += e;
    }
    sRed[row * 16 + sub] = sum;
    __syncthreads();
    if (sub == 0) {
        float ss = 0.f;
#pragma unroll
        for (int i = 0; i < 16; ++i) ss += sRed[row * 16 + i];
        sSum[row] = ss;
    }
    __syncthreads();
    float inv = 1.0f / sSum[row];
    float* arow = atn_out + ((size_t)z * LL + qb + row) * LL;
    for (int c0 = sub; c0 < LL; c0 += 16) {
        float p = srow[c0] * inv;
        srow[c0] = p;        // keep normalized probs in LDS for phase 3
        arow[c0] = p;        // atn output, written exactly once
    }
    __syncthreads();

    // ---- Phase 3: O = P V  (P from LDS, V pre-transposed in HBM) ---------
    const int ntile = wave & 3;          // which 16 of KD=64 output dims
    const int kpart = wave >> 2;         // k split: [0,1024) / [1024,2048)
    v8f c = {};
    for (int ks = 0; ks < 32; ++ks) {
        int kbase = kpart * 1024 + ks * 32;
        v16h a = load_a_frag_f32(sS + kbase, LL, lane);
        const _Float16* vb = vhT + ((size_t)z * KD + ntile * 16) * LL + kbase;
        v16h b = load_b_frag_h(vb, LL, lane);
        c = wmma_f16(a, b, c);
    }
    if (kpart == 1) {
#pragma unroll
        for (int j = 0; j < 8; ++j) sPar[ntile * 256 + j * 32 + lane] = c[j];
    }
    __syncthreads();
    if (kpart == 0) {
        int h = z >> 1, batch = z & 1;   // z = h*B + batch
#pragma unroll
        for (int j = 0; j < 8; ++j) {
            float val = c[j] + sPar[ntile * 256 + j * 32 + lane];
            size_t grow = (size_t)batch * LL + qb + j + rofs;
            int    gcol = h * KD + ntile * 16 + ncol;
            yout[grow * DIN + gcol] = (_Float16)val;   // [b, l, h*KD+d]
        }
    }
}

// ---------------------------------------------------------------------------
// LayerNorm over last dim (1024), one workgroup per row.
// ---------------------------------------------------------------------------
__global__ __launch_bounds__(256) void ln_kernel(const float* __restrict__ x,
                                                 const float* __restrict__ gamma,
                                                 const float* __restrict__ beta,
                                                 float* __restrict__ out) {
    __shared__ float red[256];
    const int t = threadIdx.x;
    const float* xr = x + (size_t)blockIdx.x * DIN;

    float s = 0.f;
    for (int c = t; c < DIN; c += 256) s += xr[c];
    red[t] = s;
    __syncthreads();
    for (int off = 128; off > 0; off >>= 1) {
        if (t < off) red[t] += red[t + off];
        __syncthreads();
    }
    float mu = red[0] * (1.0f / DIN);
    __syncthreads();

    float v = 0.f;
    for (int c = t; c < DIN; c += 256) { float d = xr[c] - mu; v += d * d; }
    red[t] = v;
    __syncthreads();
    for (int off = 128; off > 0; off >>= 1) {
        if (t < off) red[t] += red[t + off];
        __syncthreads();
    }
    float rstd = rsqrtf(red[0] * (1.0f / DIN) + 1e-5f);

    float* orow = out + (size_t)blockIdx.x * DIN;
    for (int c = t; c < DIN; c += 256)
        orow[c] = (xr[c] - mu) * rstd * gamma[c] + beta[c];
}

// ---------------------------------------------------------------------------
extern "C" void kernel_launch(void* const* d_in, const int* in_sizes, int n_in,
                              void* d_out, int out_size, void* d_ws, size_t ws_size,
                              hipStream_t stream) {
    (void)in_sizes; (void)n_in; (void)out_size; (void)ws_size;

    const float* q     = (const float*)d_in[0];
    const float* k     = (const float*)d_in[1];
    const float* v     = (const float*)d_in[2];
    const float* Wq    = (const float*)d_in[3];
    const float* bq    = (const float*)d_in[4];
    const float* Wk    = (const float*)d_in[5];
    const float* bk    = (const float*)d_in[6];
    const float* Wv    = (const float*)d_in[7];
    const float* bv    = (const float*)d_in[8];
    const float* Wo    = (const float*)d_in[9];
    const float* bo    = (const float*)d_in[10];
    const float* gamma = (const float*)d_in[11];
    const float* beta  = (const float*)d_in[12];

    float* out0 = (float*)d_out;                       // [B, L, DIN]
    float* atn  = out0 + (size_t)BB * LL * DIN;        // [H*B, L, L]

    // Workspace carve (56 MB total)
    char* w = (char*)d_ws;
    const size_t wsz = (size_t)DIN * DIN;              // 1M elements
    const size_t zld = (size_t)HH * BB * LL * KD;      // 4M elements
    _Float16* WqT = (_Float16*)w; w += wsz * sizeof(_Float16);
    _Float16* WkT = (_Float16*)w; w += wsz * sizeof(_Float16);
    _Float16* WvT = (_Float16*)w; w += wsz * sizeof(_Float16);
    _Float16* WoT = (_Float16*)w; w += wsz * sizeof(_Float16);
    _Float16* qh  = (_Float16*)w; w += zld * sizeof(_Float16);
    _Float16* kh  = (_Float16*)w; w += zld * sizeof(_Float16);
    _Float16* vhT = (_Float16*)w; w += zld * sizeof(_Float16);
    _Float16* yh  = (_Float16*)w; w += zld * sizeof(_Float16);
    float*    xb  = (float*)w;                          // pre-LN x, 16 MB

    dim3 blk(256);
    dim3 gtr(16, 16);
    wtrans_kernel<<<gtr, blk, 0, stream>>>(Wq, WqT);
    wtrans_kernel<<<gtr, blk, 0, stream>>>(Wk, WkT);
    wtrans_kernel<<<gtr, blk, 0, stream>>>(Wv, WvT);
    wtrans_kernel<<<gtr, blk, 0, stream>>>(Wo, WoT);

    dim3 ggemm(DIN / 256, MM / 64);                     // 4 x 64
    gemm_kernel<0, float><<<ggemm, blk, 0, stream>>>(q, WqT, bq, nullptr, qh, nullptr);
    gemm_kernel<0, float><<<ggemm, blk, 0, stream>>>(k, WkT, bk, nullptr, kh, nullptr);
    gemm_kernel<2, float><<<ggemm, blk, 0, stream>>>(v, WvT, bv, nullptr, vhT, nullptr);

    size_t smem = (size_t)(16 * LL + 256 + 16 + 16 + 1024) * sizeof(float); // ~136 KB
    attn_kernel<<<dim3(LL / 16, HH * BB), blk, smem, stream>>>(qh, kh, vhT, atn, yh);

    gemm_kernel<1, _Float16><<<ggemm, blk, 0, stream>>>(yh, WoT, bo, q, nullptr, xb);
    ln_kernel<<<dim3(MM), blk, 0, stream>>>(xb, gamma, beta, out0);
}
